// StTokenizer_53609781788885
// MI455X (gfx1250) — matrix operand order repaired and checked by reading, not compiled, not run on hardware
//
#include <hip/hip_runtime.h>

// ---------------- fixed problem dimensions (from reference) ----------------
#define NN   20000      // N
#define N1C  20001      // N+1 nodes
#define MC   80000      // edges
#define BC   16         // batch
#define LC   128        // seq len
#define SC   6          // dyn feature dim
#define FC   6          // static feature count
#define VC   32         // vocab
#define TC   288        // time steps (T)
#define DC   128        // model dim
#define DMC  768        // output dim
#define RDC  (BC * N1C) // 320016 rows in batched dynamic graph
#define ROWS_BL (BC * LC) // 2048

typedef __attribute__((ext_vector_type(16))) __bf16 v16bf;
typedef __attribute__((ext_vector_type(8)))  __bf16 v8bf;
typedef __attribute__((ext_vector_type(8)))  float  v8f;

// A fragment (16-bit WMMA layout): two contiguous 8-element runs 16 elems apart
__device__ __forceinline__ v16bf load_afrag(const __bf16* p) {
    v8bf lo = *(const v8bf*)p;          // global_load_b128
    v8bf hi = *(const v8bf*)(p + 16);   // global_load_b128
    v16bf a;
    #pragma unroll
    for (int j = 0; j < 8; ++j) { a[j] = lo[j]; a[j + 8] = hi[j]; }
    return a;
}
// B fragment from transposed LDS tile: two contiguous 8-element runs 16 elems apart
__device__ __forceinline__ v16bf load_bfrag(const __bf16* p) {
    v8bf lo = *(const v8bf*)p;          // ds_load_b128
    v8bf hi = *(const v8bf*)(p + 16);   // ds_load_b128
    v16bf b;
    #pragma unroll
    for (int j = 0; j < 8; ++j) { b[j] = lo[j]; b[j + 8] = hi[j]; }
    return b;
}

// ======================================================================
// WMMA GEMM:  out = act(A[M,K](bf16) @ W[K,N](bf16) + bias[N])
// One wave per 32x32 output tile: 2 A-frags x 2 B-frags -> 4 WMMAs / K-step.
// B tile staged transposed through LDS so B-frag reads are ds_load_b128.
// K must be a multiple of 32, N a multiple of 32 (true at all call sites).
// ======================================================================
__global__ __launch_bounds__(32)
void k_wmma_gemm(const __bf16* __restrict__ A, const __bf16* __restrict__ W,
                 const float* __restrict__ bias, float* __restrict__ outF,
                 __bf16* __restrict__ outB, int Mrows, int K, int N, int relu) {
    __shared__ __bf16 Bs[32 * 32];            // transposed: Bs[n*32 + k]
    const int lane = threadIdx.x;             // 0..31
    const int m0   = blockIdx.x * 32;
    const int n0   = blockIdx.y * 32;
    const int half = lane >> 4;               // 0|1
    const int l15  = lane & 15;
    const int kbA  = half ? 8 : 0;            // A frag K base (16-bit layout)
    const int kbB  = half ? 8 : 0;            // B frag K base
    // clamp rows: garbage rows only affect C rows we never store
    const int r0c = min(m0 + l15,      Mrows - 1);
    const int r1c = min(m0 + 16 + l15, Mrows - 1);
    const __bf16* a0 = A + (size_t)r0c * K;
    const __bf16* a1 = A + (size_t)r1c * K;

    v8f c00 = {}, c01 = {}, c10 = {}, c11 = {};

    for (int k0 = 0; k0 < K; k0 += 32) {
        // stage 32x32 W tile transposed: lane = row k0+lane, scatter 32 cols
        const __bf16* wrow = W + (size_t)(k0 + lane) * N + n0;
        #pragma unroll
        for (int n = 0; n < 32; ++n) Bs[n * 32 + lane] = wrow[n];
        __syncthreads();

        if (k0 + 32 < K) {
            __builtin_prefetch(a0 + k0 + 32, 0, 3);   // global_prefetch_b8
            __builtin_prefetch(a1 + k0 + 32, 0, 3);
        }
        v16bf aLo = load_afrag(a0 + k0 + kbA);
        v16bf aHi = load_afrag(a1 + k0 + kbA);
        v16bf b0  = load_bfrag(&Bs[l15 * 32 + kbB]);
        v16bf b1  = load_bfrag(&Bs[(l15 + 16) * 32 + kbB]);

        c00 = __builtin_amdgcn_wmma_f32_16x16x32_bf16(false, aLo, false, b0, (short)0, c00, false, false);
        c01 = __builtin_amdgcn_wmma_f32_16x16x32_bf16(false, aLo, false, b1, (short)0, c01, false, false);
        c10 = __builtin_amdgcn_wmma_f32_16x16x32_bf16(false, aHi, false, b0, (short)0, c10, false, false);
        c11 = __builtin_amdgcn_wmma_f32_16x16x32_bf16(false, aHi, false, b1, (short)0, c11, false, false);
        __syncthreads();
    }

    const int col0 = n0 + l15;
    const int col1 = col0 + 16;
    const float bv0 = bias ? bias[col0] : 0.f;
    const float bv1 = bias ? bias[col1] : 0.f;
    #pragma unroll
    for (int r = 0; r < 8; ++r) {
        int row0 = m0 + r + (half ? 8 : 0);
        int row1 = row0 + 16;
        float v00 = c00[r] + bv0, v01 = c01[r] + bv1;
        float v10 = c10[r] + bv0, v11 = c11[r] + bv1;
        if (relu) {
            v00 = fmaxf(v00, 0.f); v01 = fmaxf(v01, 0.f);
            v10 = fmaxf(v10, 0.f); v11 = fmaxf(v11, 0.f);
        }
        if (row0 < Mrows) {
            if (outF) { outF[(size_t)row0 * N + col0] = v00; outF[(size_t)row0 * N + col1] = v01; }
            if (outB) { outB[(size_t)row0 * N + col0] = (__bf16)v00; outB[(size_t)row0 * N + col1] = (__bf16)v01; }
        }
        if (row1 < Mrows) {
            if (outF) { outF[(size_t)row1 * N + col0] = v10; outF[(size_t)row1 * N + col1] = v11; }
            if (outB) { outB[(size_t)row1 * N + col0] = (__bf16)v10; outB[(size_t)row1 * N + col1] = (__bf16)v11; }
        }
    }
}

// ---------------- small elementwise / gather kernels ----------------
__global__ void k_cast_bf16(const float* __restrict__ in, __bf16* __restrict__ out, long n) {
    long i = (long)blockIdx.x * blockDim.x + threadIdx.x;
    if (i < n) out[i] = (__bf16)in[i];
}
__global__ void k_fill_f32(float* p, float v, long n) {
    long i = (long)blockIdx.x * blockDim.x + threadIdx.x;
    if (i < n) p[i] = v;
}
__global__ void k_fill_u32(unsigned* p, unsigned v, long n) {
    long i = (long)blockIdx.x * blockDim.x + threadIdx.x;
    if (i < n) p[i] = v;
}

// embedding gather -> se[N1, F*D] (bf16)
__global__ void k_embed(const float* __restrict__ emb, const int* __restrict__ sf,
                        __bf16* __restrict__ out) {
    long i = (long)blockIdx.x * blockDim.x + threadIdx.x;
    if (i >= (long)N1C * (FC * DC)) return;
    int n = (int)(i / (FC * DC)), c = (int)(i % (FC * DC));
    int f = c >> 7, d = c & 127;
    int v = sf[n * FC + f];
    out[i] = (__bf16)emb[((size_t)(f * VC + v) << 7) | d];
}

// es/ed = h . a_src / a_dst   (i = head*rows + node)
__global__ void k_gat_coef(const __bf16* __restrict__ h, const float* __restrict__ a_src,
                           const float* __restrict__ a_dst, float* __restrict__ es,
                           float* __restrict__ ed, int rows) {
    long i = (long)blockIdx.x * blockDim.x + threadIdx.x;
    if (i >= 2L * rows) return;
    int hh = (int)(i / rows);
    const __bf16* hp = h + (size_t)i * DC;
    float s = 0.f, dmul = 0.f;
    for (int d = 0; d < DC; ++d) {
        float hv = (float)hp[d];
        s    += hv * a_src[hh * DC + d];
        dmul += hv * a_dst[hh * DC + d];
    }
    es[i] = s; ed[i] = dmul;
}

__device__ __forceinline__ float lrelu02(float x) { return x > 0.f ? x : 0.2f * x; }
__device__ __forceinline__ unsigned ord_enc(float f) {
    int b = __float_as_int(f);
    return b < 0 ? ~(unsigned)b : ((unsigned)b | 0x80000000u);
}

// pass 1: segment max (ordered-uint atomicMax)
__global__ void k_edge_max(const int* __restrict__ edges, const float* __restrict__ es,
                           const float* __restrict__ ed, unsigned* __restrict__ mbuf,
                           int Me, int nb, int rowsTot, int strideN) {
    long i = (long)blockIdx.x * blockDim.x + threadIdx.x;
    long tot = (long)nb * Me * 2;
    if (i >= tot) return;
    int hh = (int)(i & 1);
    long p = i >> 1;
    int e = (int)(p % Me), b = (int)(p / Me);
    int src = edges[e] + b * strideN;
    int dst = edges[Me + e] + b * strideN;
    float lg = lrelu02(es[(size_t)hh * rowsTot + src] + ed[(size_t)hh * rowsTot + dst]);
    atomicMax(&mbuf[(size_t)hh * rowsTot + dst], ord_enc(lg));
}

// decode ordered max in-place to float; non-finite -> 0
__global__ void k_finmax(unsigned* m, long n) {
    long i = (long)blockIdx.x * blockDim.x + threadIdx.x;
    if (i >= n) return;
    unsigned o = m[i];
    int bits = (o & 0x80000000u) ? (int)(o & 0x7fffffffu) : (int)~o;
    float f = __int_as_float(bits);
    if (!isfinite(f)) f = 0.0f;
    ((float*)m)[i] = f;
}

// pass 2: w = exp(logit - m[dst]) * ew; den[dst] += w; keep w
__global__ void k_edge_w(const int* __restrict__ edges, const float* __restrict__ es,
                         const float* __restrict__ ed, const float* __restrict__ mfl,
                         const float* __restrict__ ew, float* __restrict__ wbuf,
                         float* __restrict__ den, int Me, int nb, int rowsTot, int strideN) {
    long i = (long)blockIdx.x * blockDim.x + threadIdx.x;
    long tot = (long)nb * Me * 2;
    if (i >= tot) return;
    int hh = (int)(i & 1);
    long p = i >> 1;
    int e = (int)(p % Me), b = (int)(p / Me);
    int src = edges[e] + b * strideN;
    int dst = edges[Me + e] + b * strideN;
    float lg = lrelu02(es[(size_t)hh * rowsTot + src] + ed[(size_t)hh * rowsTot + dst]);
    float w = __expf(lg - mfl[(size_t)hh * rowsTot + dst]) * ew[e];
    wbuf[i] = w;
    atomicAdd(&den[(size_t)hh * rowsTot + dst], w);
}

// pass 3: gout[dst] += 0.5 * alpha * h[src]   (0.5 folds the 2-head mean)
__global__ void k_edge_msg(const int* __restrict__ edges, const __bf16* __restrict__ h,
                           const float* __restrict__ wbuf, const float* __restrict__ den,
                           float* __restrict__ gout, int Me, int nb, int rowsTot, int strideN) {
    long i = (long)blockIdx.x * blockDim.x + threadIdx.x;
    long tot = (long)nb * Me * 2;
    if (i >= tot) return;
    int hh = (int)(i & 1);
    long p = i >> 1;
    int e = (int)(p % Me), b = (int)(p / Me);
    int src = edges[e] + b * strideN;
    int dst = edges[Me + e] + b * strideN;
    float alpha = 0.5f * wbuf[i] / (den[(size_t)hh * rowsTot + dst] + 1e-16f);
    const __bf16* hs = h + ((size_t)hh * rowsTot + src) * DC;
    float* gp = gout + (size_t)dst * DC;
    for (int d = 0; d < DC; ++d) atomicAdd(&gp[d], alpha * (float)hs[d]);
}

// dynamic MLP layer 0 (K=6, fp32): row r = b*N1 + n; relu(dyn[n, tid(b,0)] @ W1 + b1)
__global__ __launch_bounds__(128)
void k_dyn_mlp0(const float* __restrict__ dynf, const int* __restrict__ btid,
                const float* __restrict__ W1, const float* __restrict__ b1,
                __bf16* __restrict__ out) {
    int r = blockIdx.x, j = threadIdx.x;
    int b = r / N1C, n = r % N1C;
    int tid = btid[b * LC];
    const float* f = dynf + ((size_t)n * (TC + 1) + tid) * SC;
    float acc = b1[j];
    #pragma unroll
    for (int s = 0; s < SC; ++s) acc += f[s] * W1[s * DC + j];
    out[(size_t)r * DC + j] = (__bf16)(acc > 0.f ? acc : 0.f);
}

// road[b*L+l, 0:128]=static_emb[rid], [128:256]=dyn_emb[b*N1+rid]  (bf16)
__global__ void k_build_road(const float* __restrict__ semb, const float* __restrict__ demb,
                             const int* __restrict__ brid, __bf16* __restrict__ road) {
    long i = (long)blockIdx.x * blockDim.x + threadIdx.x;
    if (i >= (long)ROWS_BL * 256) return;
    int row = (int)(i >> 8), c = (int)(i & 255);
    int b = row / LC;
    int rid = brid[row];
    float v = (c < DC) ? semb[(size_t)rid * DC + c]
                       : demb[((size_t)b * N1C + rid) * DC + (c - DC)];
    road[i] = (__bf16)v;
}

// single-head attention per (b,l):  scores -> softmax -> @V  (d=256, L=128)
__global__ __launch_bounds__(128)
void k_attn(const float* __restrict__ q, const float* __restrict__ k,
            const float* __restrict__ v, float* __restrict__ outp) {
    int row = blockIdx.x;           // b*L + l
    int b = row >> 7;
    int t = threadIdx.x;            // 0..127
    __shared__ float qv[256], sc[128], red[128];
    qv[t]       = q[(size_t)row * 256 + t];
    qv[t + 128] = q[(size_t)row * 256 + 128 + t];
    __syncthreads();
    const float* kb = k + (size_t)b * LC * 256;
    float s = 0.f;
    for (int d = 0; d < 256; ++d) s += qv[d] * kb[(size_t)t * 256 + d];
    s *= 0.0625f;                   // 1/sqrt(256)
    sc[t] = s; red[t] = s;
    __syncthreads();
    for (int st = 64; st; st >>= 1) { if (t < st) red[t] = fmaxf(red[t], red[t + st]); __syncthreads(); }
    float mx = red[0];
    __syncthreads();
    float pexp = __expf(sc[t] - mx);
    sc[t] = pexp; red[t] = pexp;
    __syncthreads();
    for (int st = 64; st; st >>= 1) { if (t < st) red[t] += red[t + st]; __syncthreads(); }
    float inv = 1.0f / red[0];
    __syncthreads();
    const float* vb = v + (size_t)b * LC * 256;
    float o0 = 0.f, o1 = 0.f;
    for (int m2 = 0; m2 < 128; ++m2) {
        float pm = sc[m2] * inv;
        o0 += pm * vb[(size_t)m2 * 256 + t];
        o1 += pm * vb[(size_t)m2 * 256 + 128 + t];
    }
    outp[(size_t)row * 256 + t] = o0;
    outp[(size_t)row * 256 + 128 + t] = o1;
}

// attention output (fp32, 256) -> cat buffer cols [0,256) as bf16
__global__ void k_cat_road(const float* __restrict__ ao, __bf16* __restrict__ cat) {
    long i = (long)blockIdx.x * blockDim.x + threadIdx.x;
    if (i >= (long)ROWS_BL * 256) return;
    int row = (int)(i >> 8), c = (int)(i & 255);
    cat[(size_t)row * 384 + c] = (__bf16)ao[i];
}

// time projection (K=6) -> cat buffer cols [256,384)
__global__ void k_te(const float* __restrict__ btf, const float* __restrict__ Wt,
                     const float* __restrict__ bt, __bf16* __restrict__ cat) {
    long i = (long)blockIdx.x * blockDim.x + threadIdx.x;
    if (i >= (long)ROWS_BL * DC) return;
    int row = (int)(i >> 7), j = (int)(i & 127);
    const float* f = btf + (size_t)row * SC;
    float acc = bt[j];
    #pragma unroll
    for (int s = 0; s < SC; ++s) acc += f[s] * Wt[s * DC + j];
    cat[(size_t)row * 384 + 256 + j] = (__bf16)acc;
}

// ======================================================================
// host orchestration
// ======================================================================
static inline void gemm(const __bf16* A, const __bf16* W, const float* bias,
                        float* oF, __bf16* oB, int M, int K, int N, int relu,
                        hipStream_t s) {
    dim3 g((M + 31) / 32, N / 32);
    k_wmma_gemm<<<g, dim3(32), 0, s>>>(A, W, bias, oF, oB, M, K, N, relu);
}
static inline dim3 gr(long n) { return dim3((unsigned)((n + 255) / 256)); }

extern "C" void kernel_launch(void* const* d_in, const int* in_sizes, int n_in,
                              void* d_out, int out_size, void* d_ws, size_t ws_size,
                              hipStream_t stream) {
    (void)in_sizes; (void)n_in; (void)out_size; (void)ws_size;
    // --- input mapping: params leaves in JAX tree-flatten (sorted) order,
    //     then the remaining arrays in setup_inputs() insertion order ---
    const float* aWk  = (const float*)d_in[0];
    const float* aWo  = (const float*)d_in[1];
    const float* aWq  = (const float*)d_in[2];
    const float* aWv  = (const float*)d_in[3];
    const float* abk  = (const float*)d_in[4];
    const float* abo  = (const float*)d_in[5];
    const float* abq  = (const float*)d_in[6];
    const float* abv  = (const float*)d_in[7];
    const float* emb  = (const float*)d_in[8];
    const float* fW1  = (const float*)d_in[9];
    const float* fW2  = (const float*)d_in[10];
    const float* fb1  = (const float*)d_in[11];
    const float* fb2  = (const float*)d_in[12];
    const float* gdW  = (const float*)d_in[13];
    const float* gdAd = (const float*)d_in[14];
    const float* gdAs = (const float*)d_in[15];
    const float* gsW  = (const float*)d_in[16];
    const float* gsAd = (const float*)d_in[17];
    const float* gsAs = (const float*)d_in[18];
    const float* d0W1 = (const float*)d_in[19];
    const float* d0W2 = (const float*)d_in[20];
    const float* d0b1 = (const float*)d_in[21];
    const float* d0b2 = (const float*)d_in[22];
    const float* d2W1 = (const float*)d_in[23];
    const float* d2W2 = (const float*)d_in[24];
    const float* d2b1 = (const float*)d_in[25];
    const float* d2b2 = (const float*)d_in[26];
    const float* s0W1 = (const float*)d_in[27];
    const float* s0W2 = (const float*)d_in[28];
    const float* s0b1 = (const float*)d_in[29];
    const float* s0b2 = (const float*)d_in[30];
    const float* s2W1 = (const float*)d_in[31];
    const float* s2W2 = (const float*)d_in[32];
    const float* s2b1 = (const float*)d_in[33];
    const float* s2b2 = (const float*)d_in[34];
    const float* tW   = (const float*)d_in[35];
    const float* tb   = (const float*)d_in[36];
    const int*   sfeat = (const int*)d_in[37];
    const float* dynf  = (const float*)d_in[38];
    const int*   edges = (const int*)d_in[39];
    const float* ew    = (const float*)d_in[40];
    const int*   brid  = (const int*)d_in[41];
    const int*   btid  = (const int*)d_in[42];
    const float* btf   = (const float*)d_in[43];

    // --- workspace bump allocator (256B aligned) ---
    char* ws = (char*)d_ws;
    size_t cur = 0;
    auto alloc = [&](size_t bytes) -> char* {
        char* p = ws + cur;
        cur = (cur + bytes + 255) & ~(size_t)255;
        return p;
    };
    // bf16 weight pool
    __bf16* wbf = (__bf16*)alloc((size_t)966656 * 2);
    size_t w_s0W1 = 0,                 w_s0W2 = w_s0W1 + 98304;
    size_t w_gsW  = w_s0W2 + 16384,    w_s2W1 = w_gsW + 32768;
    size_t w_s2W2 = w_s2W1 + 16384,    w_d0W2 = w_s2W2 + 16384;
    size_t w_gdW  = w_d0W2 + 16384,    w_d2W1 = w_gdW + 32768;
    size_t w_d2W2 = w_d2W1 + 16384,    w_Wq   = w_d2W2 + 16384;
    size_t w_Wk   = w_Wq + 65536,      w_Wv   = w_Wk + 65536;
    size_t w_Wo   = w_Wv + 65536,      w_fW1  = w_Wo + 65536;
    size_t w_fW2  = w_fW1 + 147456;
    // static path
    __bf16* se768 = (__bf16*)alloc((size_t)N1C * 768 * 2);
    __bf16* tmpS  = (__bf16*)alloc((size_t)N1C * DC * 2);
    __bf16* xs    = (__bf16*)alloc((size_t)N1C * DC * 2);
    __bf16* hS    = (__bf16*)alloc((size_t)2 * N1C * DC * 2);
    float*  esS   = (float*)alloc((size_t)2 * N1C * 4);
    float*  edS   = (float*)alloc((size_t)2 * N1C * 4);
    unsigned* mS  = (unsigned*)alloc((size_t)2 * N1C * 4);
    float*  denS  = (float*)alloc((size_t)2 * N1C * 4);
    float*  wES   = (float*)alloc((size_t)2 * MC * 4);
    float*  goutS = (float*)alloc((size_t)N1C * DC * 4);
    float*  sembF = (float*)alloc((size_t)N1C * DC * 4);
    // dynamic path (deliberate overlays: dh1+xd region later reused as goutD/dynemb)
    char*   dynbase = alloc((size_t)RDC * DC * 4);
    __bf16* dh1   = (__bf16*)dynbase;
    __bf16* xd    = (__bf16*)(dynbase + (size_t)RDC * DC * 2);
    float*  goutD = (float*)dynbase;
    float*  dembF = (float*)dynbase;
    __bf16* hD    = (__bf16*)alloc((size_t)2 * RDC * DC * 2);
    __bf16* xd2   = hD;                       // reused after edge passes
    __bf16* dhid  = hD + (size_t)RDC * DC;    // reused after edge passes
    float*  esD   = (float*)alloc((size_t)2 * RDC * 4);
    float*  edD   = (float*)alloc((size_t)2 * RDC * 4);
    unsigned* mD  = (unsigned*)alloc((size_t)2 * RDC * 4);
    float*  denD  = (float*)alloc((size_t)2 * RDC * 4);
    float*  wED   = (float*)alloc((size_t)2 * BC * MC * 4);
    // attention / final
    __bf16* road  = (__bf16*)alloc((size_t)ROWS_BL * 256 * 2);
    float*  qF    = (float*)alloc((size_t)ROWS_BL * 256 * 4);
    float*  kF    = (float*)alloc((size_t)ROWS_BL * 256 * 4);
    float*  vF    = (float*)alloc((size_t)ROWS_BL * 256 * 4);
    float*  attnF = (float*)alloc((size_t)ROWS_BL * 256 * 4);
    float*  aoF   = (float*)alloc((size_t)ROWS_BL * 256 * 4);
    __bf16* attnB = (__bf16*)alloc((size_t)ROWS_BL * 256 * 2);
    __bf16* catB  = (__bf16*)alloc((size_t)ROWS_BL * 384 * 2);
    __bf16* finhB = (__bf16*)alloc((size_t)ROWS_BL * 384 * 2);

    const unsigned NEG_INF_ORD = 0x007FFFFFu;  // ordered encoding of -inf

    auto cast = [&](const float* src, size_t off, long n) {
        k_cast_bf16<<<gr(n), 256, 0, stream>>>(src, wbf + off, n);
    };
    // ---- weights -> bf16 ----
    cast(s0W1, w_s0W1, 98304);  cast(s0W2, w_s0W2, 16384);
    cast(gsW,  w_gsW,  32768);  cast(s2W1, w_s2W1, 16384);
    cast(s2W2, w_s2W2, 16384);  cast(d0W2, w_d0W2, 16384);
    cast(gdW,  w_gdW,  32768);  cast(d2W1, w_d2W1, 16384);
    cast(d2W2, w_d2W2, 16384);  cast(aWq,  w_Wq,   65536);
    cast(aWk,  w_Wk,   65536);  cast(aWv,  w_Wv,   65536);
    cast(aWo,  w_Wo,   65536);  cast(fW1,  w_fW1,  147456);
    cast(fW2,  w_fW2,  294912);

    // ================= static path =================
    k_embed<<<gr((long)N1C * 768), 256, 0, stream>>>(emb, sfeat, se768);
    gemm(se768, wbf + w_s0W1, s0b1, nullptr, tmpS, N1C, 768, DC, 1, stream);
    gemm(tmpS,  wbf + w_s0W2, s0b2, nullptr, xs,   N1C, DC,  DC, 0, stream);
    for (int h = 0; h < 2; ++h)
        gemm(xs, wbf + w_gsW + (size_t)h * DC * DC, nullptr, nullptr,
             hS + (size_t)h * N1C * DC, N1C, DC, DC, 0, stream);
    k_gat_coef<<<gr(2L * N1C), 256, 0, stream>>>(hS, gsAs, gsAd, esS, edS, N1C);
    k_fill_u32<<<gr(2L * N1C), 256, 0, stream>>>(mS, NEG_INF_ORD, 2L * N1C);
    k_fill_f32<<<gr(2L * N1C), 256, 0, stream>>>(denS, 0.f, 2L * N1C);
    k_fill_f32<<<gr((long)N1C * DC), 256, 0, stream>>>(goutS, 0.f, (long)N1C * DC);
    k_edge_max<<<gr(2L * MC), 256, 0, stream>>>(edges, esS, edS, mS, MC, 1, N1C, NN);
    k_finmax<<<gr(2L * N1C), 256, 0, stream>>>(mS, 2L * N1C);
    k_edge_w<<<gr(2L * MC), 256, 0, stream>>>(edges, esS, edS, (const float*)mS, ew,
                                              wES, denS, MC, 1, N1C, NN);
    k_edge_msg<<<gr(2L * MC), 256, 0, stream>>>(edges, hS, wES, denS, goutS,
                                                MC, 1, N1C, NN);
    k_cast_bf16<<<gr((long)N1C * DC), 256, 0, stream>>>(goutS, xs, (long)N1C * DC);
    gemm(xs,   wbf + w_s2W1, s2b1, nullptr, tmpS, N1C, DC, DC, 1, stream);
    gemm(tmpS, wbf + w_s2W2, s2b2, sembF,  nullptr, N1C, DC, DC, 0, stream);

    // ================= dynamic path =================
    k_dyn_mlp0<<<dim3(RDC), 128, 0, stream>>>(dynf, btid, d0W1, d0b1, dh1);
    gemm(dh1, wbf + w_d0W2, d0b2, nullptr, xd, RDC, DC, DC, 0, stream);
    for (int h = 0; h < 2; ++h)
        gemm(xd, wbf + w_gdW + (size_t)h * DC * DC, nullptr, nullptr,
             hD + (size_t)h * RDC * DC, RDC, DC, DC, 0, stream);
    k_gat_coef<<<gr(2L * RDC), 256, 0, stream>>>(hD, gdAs, gdAd, esD, edD, RDC);
    k_fill_u32<<<gr(2L * RDC), 256, 0, stream>>>(mD, NEG_INF_ORD, 2L * RDC);
    k_fill_f32<<<gr(2L * RDC), 256, 0, stream>>>(denD, 0.f, 2L * RDC);
    k_fill_f32<<<gr((long)RDC * DC), 256, 0, stream>>>(goutD, 0.f, (long)RDC * DC);
    k_edge_max<<<gr(2L * BC * MC), 256, 0, stream>>>(edges, esD, edD, mD, MC, BC, RDC, NN);
    k_finmax<<<gr(2L * RDC), 256, 0, stream>>>(mD, 2L * RDC);
    k_edge_w<<<gr(2L * BC * MC), 256, 0, stream>>>(edges, esD, edD, (const float*)mD, ew,
                                                   wED, denD, MC, BC, RDC, NN);
    k_edge_msg<<<gr(2L * BC * MC), 256, 0, stream>>>(edges, hD, wED, denD, goutD,
                                                     MC, BC, RDC, NN);
    k_cast_bf16<<<gr((long)RDC * DC), 256, 0, stream>>>(goutD, xd2, (long)RDC * DC);
    gemm(xd2,  wbf + w_d2W1, d2b1, nullptr, dhid, RDC, DC, DC, 1, stream);
    gemm(dhid, wbf + w_d2W2, d2b2, dembF, nullptr, RDC, DC, DC, 0, stream);

    // ================= attention + final =================
    k_build_road<<<gr((long)ROWS_BL * 256), 256, 0, stream>>>(sembF, dembF, brid, road);
    gemm(road, wbf + w_Wq, abq, qF, nullptr, ROWS_BL, 256, 256, 0, stream);
    gemm(road, wbf + w_Wk, abk, kF, nullptr, ROWS_BL, 256, 256, 0, stream);
    gemm(road, wbf + w_Wv, abv, vF, nullptr, ROWS_BL, 256, 256, 0, stream);
    k_attn<<<dim3(ROWS_BL), 128, 0, stream>>>(qF, kF, vF, attnF);
    k_cast_bf16<<<gr((long)ROWS_BL * 256), 256, 0, stream>>>(attnF, attnB,
                                                             (long)ROWS_BL * 256);
    gemm(attnB, wbf + w_Wo, abo, aoF, nullptr, ROWS_BL, 256, 256, 0, stream);
    k_cat_road<<<gr((long)ROWS_BL * 256), 256, 0, stream>>>(aoF, catB);
    k_te<<<gr((long)ROWS_BL * DC), 256, 0, stream>>>(btf, tW, tb, catB);
    gemm(catB,  wbf + w_fW1, fb1, nullptr, finhB, ROWS_BL, 384, 384, 1, stream);
    gemm(finhB, wbf + w_fW2, fb2, (float*)d_out, nullptr, ROWS_BL, 384, DMC, 0, stream);
}
